// BertCrossAttention_12163347383204
// MI455X (gfx1250) — matrix operand order repaired and checked by reading, not compiled, not run on hardware
//
#include <hip/hip_runtime.h>
#include <hip/hip_bf16.h>

// ---------------------------------------------------------------------------
// BERT cross-attention layer for MI455X (gfx1250)
// bf16 WMMA + fp32 accumulate, TDM (tensor_load_to_lds) double-buffered GEMM
// ---------------------------------------------------------------------------

typedef __attribute__((ext_vector_type(16))) __bf16 v16bf;
typedef __attribute__((ext_vector_type(8)))  __bf16 v8bf;
typedef __attribute__((ext_vector_type(8)))  float  v8f;
typedef __attribute__((ext_vector_type(4))) unsigned int u32x4;
typedef __attribute__((ext_vector_type(8))) int i32x8;
typedef __attribute__((ext_vector_type(4))) int i32x4;

#define D_MODEL 1024
#define N_HEADS 16
#define HEAD_D  64
#define SEQ     512
#define BATCH   8
#define ROWS    (BATCH * SEQ)   // 4096
#define LDS_LD  40               // padded LDS row: 32 bf16 + 8 pad (80B stride)

static __device__ __forceinline__ v8f wmma_bf16f32(v16bf a, v16bf b, v8f c) {
  return __builtin_amdgcn_wmma_f32_16x16x32_bf16(false, a, false, b, (short)0, c,
                                                 false, false);
}

// A-operand gather (16x32 bf16): lanes 0-15 (M=lane) hold K={0..7,16..23},
// lanes 16-31 hold K={8..15,24..31} -> two contiguous 16B chunks per lane.
static __device__ __forceinline__ v16bf
load_a16x32(const __bf16* __restrict__ base, int ld, int row0, int k0) {
  const int lane = threadIdx.x & 31;
  const int hl = lane >> 4, r = lane & 15;
  const __bf16* p = base + (size_t)(row0 + r) * ld + k0 + hl * 8;
  v8bf lo = *(const v8bf*)(p);
  v8bf hi = *(const v8bf*)(p + 16);
  v16bf a;
#pragma unroll
  for (int j = 0; j < 8; ++j) { a[j] = lo[j]; a[j + 8] = hi[j]; }
  return a;
}

// B-operand gather (32x16 bf16) from BT[N][K] storage: lane n = lane&15,
// lanes 0-15 hold K={0..7,16..23}, lanes 16-31 hold K={8..15,24..31} -> two 16B chunks.
static __device__ __forceinline__ v16bf
load_b32x16(const __bf16* __restrict__ bt, int ld, int n0, int k0) {
  const int lane = threadIdx.x & 31;
  const int hl = lane >> 4, r = lane & 15;
  const __bf16* p = bt + (size_t)(n0 + r) * ld + k0 + hl * 8;
  v8bf lo = *(const v8bf*)(p);
  v8bf hi = *(const v8bf*)(p + 16);
  v16bf b;
#pragma unroll
  for (int j = 0; j < 8; ++j) { b[j] = lo[j]; b[j + 8] = hi[j]; }
  return b;
}

// ---------------------------------------------------------------------------
// TDM descriptor helpers (CDNA5 ISA)
// ---------------------------------------------------------------------------
static __device__ __forceinline__ unsigned lds_addr_of(const void* p) {
  // generic -> LDS address space; ptrtoint of AS(3) pointer = LDS byte offset
  return (unsigned)(uintptr_t)(__attribute__((address_space(3))) const void*)p;
}

static __device__ __forceinline__ u32x4 tdm_g0(unsigned lds, unsigned long long ga) {
  u32x4 g0;
  g0[0] = 1u;                                        // count=1 (valid user D#)
  g0[1] = lds;                                       // lds_addr (bytes)
  g0[2] = (unsigned)(ga & 0xffffffffu);              // global_addr[31:0]
  g0[3] = (unsigned)((ga >> 32) & 0x1ffffffu)        // global_addr[56:32]
        | (2u << 30);                                // type = 2 ("image")
  return g0;
}

// 2D tile: data_size=2B, padding 16B per 64B (LDS row stride 80B)
static __device__ __forceinline__ i32x8 tdm_g1(unsigned tensor_d0, unsigned tensor_d1,
                                               unsigned tile_d0, unsigned tile_d1,
                                               unsigned stride0) {
  i32x8 g1;
  g1[0] = (1 << 16)                                  // data_size = 1 -> 2 bytes
        | (1 << 20)                                  // pad_enable
        | (3 << 22)                                  // pad_interval: every 16 dw (64B)
        | (3 << 25);                                 // pad_amount: 4 dwords (16B)
  g1[1] = (int)((tensor_d0 & 0xffffu) << 16);        // tensor_dim0[15:0]
  g1[2] = (int)(((tensor_d0 >> 16) & 0xffffu) | ((tensor_d1 & 0xffffu) << 16));
  g1[3] = (int)(((tensor_d1 >> 16) & 0xffffu) | ((tile_d0 & 0xffffu) << 16));
  g1[4] = (int)(tile_d1 & 0xffffu);                  // tile_dim1 (tile_dim2 = 0)
  g1[5] = (int)stride0;                              // tensor_dim0_stride[31:0]
  g1[6] = 0;
  g1[7] = 0;
  return g1;
}

// ---------------------------------------------------------------------------
// Kernel 1: fp32 -> bf16 flat convert
// ---------------------------------------------------------------------------
__global__ void k_f32_to_bf16(const float* __restrict__ in,
                              __bf16* __restrict__ out, int n) {
  int i = blockIdx.x * blockDim.x + threadIdx.x;
  if (i < n) out[i] = (__bf16)in[i];
}

// ---------------------------------------------------------------------------
// Kernel 2: W[k][n] -> WT[n][k] as bf16 (LDS-tiled transpose)
// ---------------------------------------------------------------------------
__global__ void k_transpose_w_bf16(const float* __restrict__ W,
                                   __bf16* __restrict__ WT) {
  __shared__ float tile[16][17];
  const int kb = blockIdx.x * 16, nb = blockIdx.y * 16;
  const int tx = threadIdx.x, ty = threadIdx.y;
  tile[ty][tx] = W[(size_t)(kb + ty) * D_MODEL + nb + tx];
  __syncthreads();
  WT[(size_t)(nb + ty) * D_MODEL + kb + tx] = (__bf16)tile[tx][ty];
}

// ---------------------------------------------------------------------------
// Kernel 3: GEMM  C[m][n] = sum_k A[m][k] * BT[n][k] + bias[n]
// WG = 128 threads (4 waves), tile 32x256; per k-step TDM stages
// A-panel 32x32 and B-panel 256x32 into padded LDS, double buffered.
//   mode 0: out bf16 head-split [B,H,S,64]
//   mode 1: out bf16 head-split transposed [B,H,64,S]   (for V)
//   mode 2: out f32 plain [M,1024]                      (output projection)
// ---------------------------------------------------------------------------
__global__ void __launch_bounds__(128)
k_gemm_bf16(const __bf16* __restrict__ A, const __bf16* __restrict__ BT,
            const float* __restrict__ bias, void* __restrict__ out, int mode) {
  __shared__ __bf16 sA[2][32 * LDS_LD];    //  2 x 2.5 KB
  __shared__ __bf16 sB[2][256 * LDS_LD];   //  2 x 20  KB

  const int lane = threadIdx.x & 31;
  const int wave = threadIdx.x >> 5;
  const int m0 = blockIdx.x * 32;
  const int nblk = blockIdx.y * 256;
  const int n0w = wave * 64;

  const unsigned long long gA = (unsigned long long)(uintptr_t)A;
  const unsigned long long gB = (unsigned long long)(uintptr_t)BT;

  const i32x4 z4 = {0, 0, 0, 0};
  const i32x8 z8 = {0, 0, 0, 0, 0, 0, 0, 0};
  const i32x8 g1A = tdm_g1(D_MODEL, ROWS,    32, 32,  D_MODEL);
  const i32x8 g1B = tdm_g1(D_MODEL, D_MODEL, 32, 256, D_MODEL);

  // prologue: stage k-step 0 into buffer 0
  if (wave == 0) {
    __builtin_amdgcn_tensor_load_to_lds(
        tdm_g0(lds_addr_of(&sA[0][0]), gA + 2ull * ((size_t)m0 * D_MODEL)),
        g1A, z4, z4, z8, 0);
    __builtin_amdgcn_tensor_load_to_lds(
        tdm_g0(lds_addr_of(&sB[0][0]), gB + 2ull * ((size_t)nblk * D_MODEL)),
        g1B, z4, z4, z8, 0);
  }

  v8f acc[2][4] = {};
  for (int t = 0; t < 32; ++t) {
    const int buf = t & 1;
    if (wave == 0) {
      if (t < 31) {
        const int k1 = (t + 1) * 32;
        __builtin_amdgcn_tensor_load_to_lds(
            tdm_g0(lds_addr_of(&sA[buf ^ 1][0]),
                   gA + 2ull * ((size_t)m0 * D_MODEL + k1)),
            g1A, z4, z4, z8, 0);
        __builtin_amdgcn_tensor_load_to_lds(
            tdm_g0(lds_addr_of(&sB[buf ^ 1][0]),
                   gB + 2ull * ((size_t)nblk * D_MODEL + k1)),
            g1B, z4, z4, z8, 0);
        __builtin_amdgcn_s_wait_tensorcnt(2);   // tile t has landed
      } else {
        __builtin_amdgcn_s_wait_tensorcnt(0);
      }
    }
    __syncthreads();

    v16bf breg[4];
#pragma unroll
    for (int nt = 0; nt < 4; ++nt)
      breg[nt] = load_b32x16(&sB[buf][0], LDS_LD, n0w + nt * 16, 0);
    v16bf a0 = load_a16x32(&sA[buf][0], LDS_LD, 0, 0);
    v16bf a1 = load_a16x32(&sA[buf][0], LDS_LD, 16, 0);
#pragma unroll
    for (int nt = 0; nt < 4; ++nt) {
      acc[0][nt] = wmma_bf16f32(a0, breg[nt], acc[0][nt]);
      acc[1][nt] = wmma_bf16f32(a1, breg[nt], acc[1][nt]);
    }
    __syncthreads();
  }

  const int hl = lane >> 4, r = lane & 15;
#pragma unroll
  for (int mt = 0; mt < 2; ++mt) {
#pragma unroll
    for (int nt = 0; nt < 4; ++nt) {
#pragma unroll
      for (int v = 0; v < 8; ++v) {
        const int m = m0 + mt * 16 + hl * 8 + v;
        const int n = nblk + n0w + nt * 16 + r;
        const float val = acc[mt][nt][v] + bias[n];
        if (mode == 2) {
          ((float*)out)[(size_t)m * D_MODEL + n] = val;
        } else {
          const int b = m >> 9, s = m & 511;
          const int h = n >> 6, d = n & 63;
          size_t idx;
          if (mode == 0)        // [B,H,S,64]
            idx = (((size_t)(b * N_HEADS + h) * SEQ) + s) * HEAD_D + d;
          else                  // [B,H,64,S]
            idx = (((size_t)(b * N_HEADS + h) * HEAD_D) + d) * SEQ + s;
          ((__bf16*)out)[idx] = (__bf16)val;
        }
      }
    }
  }
}

// ---------------------------------------------------------------------------
// Kernel 4: attention core.  One 32-thread WG (1 wave) per (b, h, 16-query
// tile).  Scores staged f32 in LDS, double softmax (cl_att=1), ctx via WMMA.
// ---------------------------------------------------------------------------
__global__ void __launch_bounds__(32)
k_attention(const __bf16* __restrict__ Qb, const __bf16* __restrict__ Kb,
            const __bf16* __restrict__ VTb, const float* __restrict__ mask,
            __bf16* __restrict__ Ctx) {
  __shared__ float mLds[SEQ];          // 2 KB
  __shared__ float sc[16 * SEQ];       // 32 KB scores/probs

  const int bh = blockIdx.x;           // 0..127
  const int b = bh >> 4, h = bh & 15;
  const int q0 = blockIdx.y * 16;
  const int lane = threadIdx.x & 31;
  const int hl = lane >> 4, r = lane & 15;

  for (int i = threadIdx.x; i < SEQ; i += 32) mLds[i] = mask[b * SEQ + i];
  __syncthreads();

  const __bf16* Qh = Qb + (size_t)bh * SEQ * HEAD_D;
  const __bf16* Kh = Kb + (size_t)bh * SEQ * HEAD_D;
  const __bf16* Vh = VTb + (size_t)bh * HEAD_D * SEQ;

  // ---- scores = (Q Kᵀ)/8 + mask ----
  v16bf aq0 = load_a16x32(Qh, HEAD_D, q0, 0);
  v16bf aq1 = load_a16x32(Qh, HEAD_D, q0, 32);
  for (int kt = 0; kt < 32; ++kt) {
    const int kc = kt * 16;
    v16bf b0 = load_b32x16(Kh, HEAD_D, kc, 0);
    v16bf b1 = load_b32x16(Kh, HEAD_D, kc, 32);
    v8f c = {};
    c = wmma_bf16f32(aq0, b0, c);
    c = wmma_bf16f32(aq1, b1, c);
#pragma unroll
    for (int v = 0; v < 8; ++v) {
      const int col = kc + r;
      sc[(hl * 8 + v) * SEQ + col] = c[v] * 0.125f + mLds[col];
    }
  }
  __syncthreads();

  // ---- double softmax: lanes (l, l^16) pair on row l&15, 256 cols each ----
  {
    const int row = lane & 15;
    const int cb = (lane >> 4) * 256;
    float* prow = sc + row * SEQ;

    float mx = -3.0e38f;
    for (int c = 0; c < 256; ++c) mx = fmaxf(mx, prow[cb + c]);
    mx = fmaxf(mx, __shfl_xor(mx, 16, 32));
    float sum = 0.f;
    for (int c = 0; c < 256; ++c) {
      float e = __expf(prow[cb + c] - mx);
      prow[cb + c] = e;
      sum += e;
    }
    sum += __shfl_xor(sum, 16, 32);
    const float inv = 1.0f / sum;

    float mx2 = -3.0e38f;
    for (int c = 0; c < 256; ++c) {
      float v2 = 1.0f - prow[cb + c] * inv + mLds[cb + c];
      mx2 = fmaxf(mx2, v2);
    }
    mx2 = fmaxf(mx2, __shfl_xor(mx2, 16, 32));
    float sum2 = 0.f;
    for (int c = 0; c < 256; ++c) {
      float v2 = 1.0f - prow[cb + c] * inv + mLds[cb + c];
      float e = __expf(v2 - mx2);
      prow[cb + c] = e;
      sum2 += e;
    }
    sum2 += __shfl_xor(sum2, 16, 32);
    const float inv2 = 1.0f / sum2;
    for (int c = 0; c < 256; ++c) prow[cb + c] *= inv2;
  }
  __syncthreads();

  // ---- ctx = probs @ V ----
  v8f acc[4] = {};
  for (int ks = 0; ks < 16; ++ks) {
    const int k0 = ks * 32;
    const float* pr = sc + r * SEQ;
    v16bf ap;
#pragma unroll
    for (int j = 0; j < 16; ++j) {
      const int kk = k0 + j + ((j >= 8) ? 8 : 0) + hl * 8;
      ap[j] = (__bf16)pr[kk];
    }
#pragma unroll
    for (int nt = 0; nt < 4; ++nt) {
      v16bf bv = load_b32x16(Vh, SEQ, nt * 16, k0);
      acc[nt] = wmma_bf16f32(ap, bv, acc[nt]);
    }
  }

#pragma unroll
  for (int nt = 0; nt < 4; ++nt) {
#pragma unroll
    for (int v = 0; v < 8; ++v) {
      const int q = q0 + hl * 8 + v;
      const int d = nt * 16 + r;
      Ctx[(size_t)(b * SEQ + q) * D_MODEL + h * HEAD_D + d] = (__bf16)acc[nt][v];
    }
  }
}

// ---------------------------------------------------------------------------
// Kernel 5: residual + LayerNorm (fp32).  One 256-thread WG per row.
// ---------------------------------------------------------------------------
__global__ void k_residual_ln(const float* __restrict__ Hf,
                              const float* __restrict__ X1,
                              const float* __restrict__ lnw,
                              const float* __restrict__ lnb,
                              float* __restrict__ out) {
  __shared__ float red[256], red2[256];
  const int row = blockIdx.x;
  const float* hrow = Hf + (size_t)row * D_MODEL;
  const float* xrow = X1 + (size_t)row * D_MODEL;
  const int t = threadIdx.x;

  float vals[4], s = 0.f, s2 = 0.f;
#pragma unroll
  for (int i = 0; i < 4; ++i) {
    const int c = t + i * 256;
    const float v = hrow[c] + xrow[c];
    vals[i] = v;
    s += v;
    s2 += v * v;
  }
  red[t] = s; red2[t] = s2;
  __syncthreads();
  for (int off = 128; off > 0; off >>= 1) {
    if (t < off) { red[t] += red[t + off]; red2[t] += red2[t + off]; }
    __syncthreads();
  }
  const float mean = red[0] * (1.0f / D_MODEL);
  const float var  = red2[0] * (1.0f / D_MODEL) - mean * mean;
  const float rstd = rsqrtf(var + 1e-12f);
#pragma unroll
  for (int i = 0; i < 4; ++i) {
    const int c = t + i * 256;
    out[(size_t)row * D_MODEL + c] = lnw[c] * ((vals[i] - mean) * rstd) + lnb[c];
  }
}

// ---------------------------------------------------------------------------
// Host: orchestrate the pipeline on `stream`
// ---------------------------------------------------------------------------
extern "C" void kernel_launch(void* const* d_in, const int* in_sizes, int n_in,
                              void* d_out, int out_size, void* d_ws, size_t ws_size,
                              hipStream_t stream) {
  (void)in_sizes; (void)n_in; (void)out_size; (void)ws_size;

  const float* X1   = (const float*)d_in[0];
  const float* X2   = (const float*)d_in[1];
  const float* Mask = (const float*)d_in[2];
  const float* Wq   = (const float*)d_in[3];
  const float* bq   = (const float*)d_in[4];
  const float* Wk   = (const float*)d_in[5];
  const float* bk   = (const float*)d_in[6];
  const float* Wv   = (const float*)d_in[7];
  const float* bv   = (const float*)d_in[8];
  const float* Wo   = (const float*)d_in[9];
  const float* bo   = (const float*)d_in[10];
  const float* lnw  = (const float*)d_in[11];
  const float* lnb  = (const float*)d_in[12];
  float* Out = (float*)d_out;

  char* ws = (char*)d_ws;
  const size_t XB = (size_t)ROWS * D_MODEL * sizeof(__bf16);      // 8 MB
  const size_t WB = (size_t)D_MODEL * D_MODEL * sizeof(__bf16);   // 2 MB
  __bf16* Xb1 = (__bf16*)(ws);              size_t off = XB;
  __bf16* Xb2 = (__bf16*)(ws + off);        off += XB;
  __bf16* WqT = (__bf16*)(ws + off);        off += WB;
  __bf16* WkT = (__bf16*)(ws + off);        off += WB;
  __bf16* WvT = (__bf16*)(ws + off);        off += WB;
  __bf16* WoT = (__bf16*)(ws + off);        off += WB;
  __bf16* Qb  = (__bf16*)(ws + off);        off += XB;
  __bf16* Kb  = (__bf16*)(ws + off);        off += XB;
  __bf16* VTb = (__bf16*)(ws + off);        off += XB;
  __bf16* Ctx = (__bf16*)(ws + off);        off += XB;
  float*  Hf  = (float*)(ws + off);         // 16 MB

  const int nElem = ROWS * D_MODEL;

  k_f32_to_bf16<<<(nElem + 255) / 256, 256, 0, stream>>>(X1, Xb1, nElem);
  k_f32_to_bf16<<<(nElem + 255) / 256, 256, 0, stream>>>(X2, Xb2, nElem);

  dim3 tg(D_MODEL / 16, D_MODEL / 16), tb(16, 16);
  k_transpose_w_bf16<<<tg, tb, 0, stream>>>(Wq, WqT);
  k_transpose_w_bf16<<<tg, tb, 0, stream>>>(Wk, WkT);
  k_transpose_w_bf16<<<tg, tb, 0, stream>>>(Wv, WvT);
  k_transpose_w_bf16<<<tg, tb, 0, stream>>>(Wo, WoT);

  dim3 gg(ROWS / 32, D_MODEL / 256);      // 128 x 4 workgroups
  k_gemm_bf16<<<gg, 128, 0, stream>>>(Xb1, WqT, bq, (void*)Qb, 0);
  k_gemm_bf16<<<gg, 128, 0, stream>>>(Xb2, WkT, bk, (void*)Kb, 0);
  k_gemm_bf16<<<gg, 128, 0, stream>>>(Xb2, WvT, bv, (void*)VTb, 1);

  k_attention<<<dim3(BATCH * N_HEADS, SEQ / 16), 32, 0, stream>>>(
      Qb, Kb, VTb, Mask, Ctx);

  k_gemm_bf16<<<gg, 128, 0, stream>>>(Ctx, WoT, bo, (void*)Hf, 2);

  k_residual_ln<<<ROWS, 256, 0, stream>>>(Hf, X1, lnw, lnb, Out);
}
